// Qwen3Attention_14018773254802
// MI455X (gfx1250) — matrix operand order repaired and checked by reading, not compiled, not run on hardware
//
#include <hip/hip_runtime.h>
#include <hip/hip_bf16.h>
#include <math.h>

// ---------------- problem constants ----------------
#define S_LEN 4096
#define HID   2048
#define NH    16
#define NKV   8
#define HD    128
#define EPS   1e-6f

// ---------------- WMMA vector types ----------------
typedef __bf16 v16bf __attribute__((ext_vector_type(16)));
typedef float  v8f   __attribute__((ext_vector_type(8)));
typedef unsigned int u32x4 __attribute__((ext_vector_type(4)));
typedef int i32x4 __attribute__((ext_vector_type(4)));
typedef int i32x8 __attribute__((ext_vector_type(8)));

union v16bf_u { v16bf v; __bf16 e[16]; };
union v8f_u   { v8f   v; float  e[8];  };

struct alignas(16) bf16x8 { __bf16 e[8]; };

// ---------------- CDNA5 feature probes (compile-safe) ----------------
#if defined(__has_builtin)
#if __has_builtin(__builtin_amdgcn_global_load_async_to_lds_b128) && \
    __has_builtin(__builtin_amdgcn_s_wait_asynccnt)
#define HAVE_ASYNC 1
#endif
#if __has_builtin(__builtin_amdgcn_tensor_load_to_lds) && \
    __has_builtin(__builtin_amdgcn_s_wait_tensorcnt)
#define HAVE_TDM 1
#endif
#endif
#ifndef HAVE_ASYNC
#define HAVE_ASYNC 0
#endif
#ifndef HAVE_TDM
#define HAVE_TDM 0
#endif

#if HAVE_ASYNC
// global -> LDS direct async copy, 16B per lane (ASYNCcnt-tracked).
// Builtin signature (from hipcc diagnostic): (v4i addrspace(1)*, v4i addrspace(3)*, imm, imm)
#define ASYNC_B128(gsrc, ldst)                                                     \
    __builtin_amdgcn_global_load_async_to_lds_b128(                                \
        (__attribute__((address_space(1))) i32x4*)(uintptr_t)(gsrc),               \
        (__attribute__((address_space(3))) i32x4*)(unsigned int)(uintptr_t)(ldst), \
        0, 0)
#endif

static __device__ __forceinline__ v8f wmma_bf16(v16bf a, v16bf b, v8f c) {
    return __builtin_amdgcn_wmma_f32_16x16x32_bf16(false, a, false, b, (short)0, c, false, false);
}

// ---------------- fp32 -> bf16 convert ----------------
__global__ __launch_bounds__(256) void f32_to_bf16(const float* __restrict__ in,
                                                   __bf16* __restrict__ out, int n) {
    int i = blockIdx.x * 256 + threadIdx.x;
    if (i < n) out[i] = (__bf16)in[i];
}

// ---------------- RMSNorm over hidden (H=2048), out bf16 ----------------
__global__ __launch_bounds__(256) void rmsnorm_x(const float* __restrict__ x,
                                                 const float* __restrict__ wn,
                                                 __bf16* __restrict__ out) {
    __shared__ float red[256];
    int s = blockIdx.x, t = threadIdx.x;
    const float* row = x + (size_t)s * HID;
    float acc = 0.f;
#pragma unroll
    for (int j = 0; j < 8; ++j) { float v = row[t + 256 * j]; acc += v * v; }
    red[t] = acc; __syncthreads();
    for (int off = 128; off > 0; off >>= 1) {
        if (t < off) red[t] += red[t + off];
        __syncthreads();
    }
    float rms = rsqrtf(red[0] / (float)HID + EPS);
#pragma unroll
    for (int j = 0; j < 8; ++j) {
        int c = t + 256 * j;
        out[(size_t)s * HID + c] = (__bf16)(row[c] * rms * wn[c]);
    }
}

// ---------------- tiled bf16 WMMA GEMM: C[M,N] = A[M,K] * B[N,K]^T ----------------
// block tile 128x128, K-step 32; 256 threads = 8 waves; wave w -> rows w*16..w*16+15,
// 8 WMMA n-tiles per wave per K-step. Double-buffered LDS: async-copy tile k+1
// while WMMAs consume tile k; wait only at iteration end.
#define GBM 128
#define GBN 128
#define GBK 32
#define APAD 40   // padded LDS row stride (bf16 elems) to break bank conflicts

__global__ __launch_bounds__(256) void gemm_bf16_f32(const __bf16* __restrict__ A,
                                                     const __bf16* __restrict__ B,
                                                     float* __restrict__ C,
                                                     int M, int N, int K) {
    __shared__ __bf16 As[2][GBM][APAD];
    __shared__ __bf16 Bs[2][GBN][APAD];
    int tid = threadIdx.x;
    int w = tid >> 5, lane = tid & 31;
    int g = lane >> 4, ln = lane & 15;
    int bm = blockIdx.y * GBM, bn = blockIdx.x * GBN;

    auto stage = [&](int k0, int buf) {
#pragma unroll
        for (int j = 0; j < 2; ++j) {
            int idx = tid + 256 * j;
            int row = idx >> 2, cg = idx & 3;
#if HAVE_ASYNC
            ASYNC_B128(&A[(size_t)(bm + row) * K + k0 + cg * 8], &As[buf][row][cg * 8]);
            ASYNC_B128(&B[(size_t)(bn + row) * K + k0 + cg * 8], &Bs[buf][row][cg * 8]);
#else
            *(bf16x8*)&As[buf][row][cg * 8] =
                *(const bf16x8*)&A[(size_t)(bm + row) * K + k0 + cg * 8];
            *(bf16x8*)&Bs[buf][row][cg * 8] =
                *(const bf16x8*)&B[(size_t)(bn + row) * K + k0 + cg * 8];
#endif
        }
    };

    v8f_u acc[8] = {};

    // prologue: stage tile 0
    stage(0, 0);
#if HAVE_ASYNC
    __builtin_amdgcn_s_wait_asynccnt(0);
#endif
    __syncthreads();

    int cur = 0;
    for (int k0 = 0; k0 < K; k0 += GBK) {
        bool more = (k0 + GBK) < K;
        if (more) stage(k0 + GBK, cur ^ 1);   // overlap DMA with WMMA below

        v16bf_u a, b[8];
#pragma unroll
        for (int i = 0; i < 16; ++i)
            a.e[i] = As[cur][w * 16 + ln][16 * g + i];
#pragma unroll
        for (int t = 0; t < 8; ++t)
#pragma unroll
            for (int i = 0; i < 16; ++i)
                b[t].e[i] = Bs[cur][t * 16 + ln][16 * g + i];
#pragma unroll
        for (int t = 0; t < 8; ++t)
            acc[t].v = wmma_bf16(a.v, b[t].v, acc[t].v);

        if (more) {
#if HAVE_ASYNC
            __builtin_amdgcn_s_wait_asynccnt(0);
#endif
            __syncthreads();
            cur ^= 1;
        }
    }
#pragma unroll
    for (int t = 0; t < 8; ++t)
#pragma unroll
        for (int r = 0; r < 8; ++r)
            C[(size_t)(bm + w * 16 + r + 8 * g) * N + bn + t * 16 + ln] = acc[t].e[r];
}

// ---------------- per-head RMSNorm + RoPE, out bf16 ----------------
__global__ __launch_bounds__(128) void qknorm_rope(const float* __restrict__ in,
                                                   __bf16* __restrict__ out,
                                                   const float* __restrict__ normw,
                                                   const float* __restrict__ cosb,
                                                   const float* __restrict__ sinb,
                                                   int nheads) {
    __shared__ float xs[128];
    __shared__ float red[128];
    int s = blockIdx.x, hh = blockIdx.y, d = threadIdx.x;
    size_t base = (size_t)s * nheads * HD + (size_t)hh * HD;
    float x = in[base + d];
    red[d] = x * x; __syncthreads();
    for (int off = 64; off > 0; off >>= 1) {
        if (d < off) red[d] += red[d + off];
        __syncthreads();
    }
    float rms = rsqrtf(red[0] / (float)HD + EPS);
    float xn = x * rms * normw[d];
    xs[d] = xn; __syncthreads();
    float rot = (d < 64) ? -xs[d + 64] : xs[d - 64];
    float val = xn * cosb[(size_t)s * HD + d] + rot * sinb[(size_t)s * HD + d];
    out[base + d] = (__bf16)val;
}

// ---------------- causal GQA flash attention ----------------
// grid = (S/64, NH); block = 128 threads = 4 waves; wave w owns 16 query rows.
// Double-buffered K tile via TDM (padded DMA directly into LDS) + V tile
// (transposed in LDS). TDM/loads for block k+1 issued before the WMMA/softmax
// work on block k; s_wait_tensorcnt deferred to iteration end.
#define KPAD 136   // 128 data + 8 pad bf16 (= TDM pad_interval 64dw / pad_amount 4dw)
#define VPAD 72
#define PPAD 72

__global__ __launch_bounds__(128) void flash_attn(const __bf16* __restrict__ qb,
                                                  const __bf16* __restrict__ kb,
                                                  const __bf16* __restrict__ vb,
                                                  __bf16* __restrict__ ob) {
    __shared__ __bf16 Ks[2][64][KPAD];
    __shared__ __bf16 Vt[2][128][VPAD];
    __shared__ __bf16 Ps[4][16][PPAD];

    int tid = threadIdx.x;
    int w = tid >> 5, lane = tid & 31;
    int g = lane >> 4, ln = lane & 15;
    int q0 = blockIdx.x * 64;
    int h = blockIdx.y;
    int hkv = h >> 1;            // rep = NH/NKV = 2

    auto stage_kv = [&](int kblk2, int buf) {
#if HAVE_TDM
        // ---- K tile: one TDM descriptor DMAs 64x128 bf16 into LDS with padding ----
        if (w == 0) {
            unsigned ldsa = (unsigned)(uintptr_t)(&Ks[buf][0][0]);
            unsigned long long ga = (unsigned long long)(uintptr_t)kb +
                2ull * ((size_t)(kblk2 * 64) * (NKV * HD) + (size_t)hkv * HD);
            u32x4 g0;
            g0.x = 1u;                                    // count=1, user mode
            g0.y = ldsa;                                  // lds_addr
            g0.z = (unsigned)(ga & 0xffffffffu);          // global_addr[31:0]
            g0.w = (unsigned)((ga >> 32) & 0x1ffffffu)    // global_addr[56:32]
                 | (2u << 30);                            // type=2 (image)
            i32x8 g1;
            g1.s0 = (1 << 16)        // data_size = 2 bytes
                  | (1 << 20)        // pad_enable
                  | (5 << 22)        // pad_interval: 64 dwords (= one 128-elem row)
                  | (3 << 25);       // pad_amount: 4 dwords (= 8 bf16 -> KPAD 136)
            g1.s1 = (int)(128u << 16);     // tensor_dim0 = 128 (bits 79:48, low half)
            g1.s2 = (int)(0x1000u << 16);  // tensor_dim1 = 4096 (bits 111:80, low half)
            g1.s3 = (int)(128u << 16);     // tile_dim0 = 128 (bits 127:112)
            g1.s4 = 64;                    // tile_dim1 = 64, tile_dim2 = 0
            g1.s5 = NKV * HD;              // tensor_dim0_stride = 1024 (low 32)
            g1.s6 = 0;
            g1.s7 = 0;
            i32x4 gz = {0, 0, 0, 0};
#if __clang_major__ >= 23
            i32x8 gz8 = {0, 0, 0, 0, 0, 0, 0, 0};
            __builtin_amdgcn_tensor_load_to_lds(g0, g1, gz, gz, gz8, 0);
#else
            __builtin_amdgcn_tensor_load_to_lds(g0, g1, gz, gz, 0);
#endif
        }
#endif
        // ---- V tile (transposed into LDS) + K fallback ----
#pragma unroll
        for (int j = 0; j < 8; ++j) {
            int idx = tid + 128 * j;
            int row = idx >> 4, grp = idx & 15;
            size_t src = (size_t)(kblk2 * 64 + row) * (NKV * HD) + (size_t)hkv * HD + grp * 8;
#if !HAVE_TDM
            *(bf16x8*)&Ks[buf][row][grp * 8] = *(const bf16x8*)&kb[src];
#endif
            bf16x8 vv = *(const bf16x8*)&vb[src];
#pragma unroll
            for (int e = 0; e < 8; ++e) Vt[buf][grp * 8 + e][row] = vv.e[e];
        }
    };

    // Q fragments for the full D=128 (4 K-chunks of 32)
    v16bf_u aq[4];
    {
        int qrow = q0 + w * 16 + ln;
        const __bf16* qp = qb + (size_t)qrow * (NH * HD) + (size_t)h * HD;
#pragma unroll
        for (int c = 0; c < 4; ++c)
#pragma unroll
            for (int i = 0; i < 16; ++i)
                aq[c].e[i] = qp[c * 32 + 16 * g + i];
    }

    v8f_u o[8] = {};
    float mst[8], lst[8];
#pragma unroll
    for (int r = 0; r < 8; ++r) { mst[r] = -1e30f; lst[r] = 0.f; }

    const float scale = 0.08838834764831845f;   // 1/sqrt(128)
    int kbmax = (q0 + 63) / 64;                 // inclusive causal limit

    // prologue: stage key block 0
    stage_kv(0, 0);
#if HAVE_TDM
    if (w == 0) __builtin_amdgcn_s_wait_tensorcnt(0);
#endif
    __syncthreads();

    int cur = 0;
    for (int kblk = 0; kblk <= kbmax; ++kblk) {
        bool more = kblk < kbmax;
        if (more) stage_kv(kblk + 1, cur ^ 1);   // overlap TDM/loads with math below

        // S = Q * K^T  (16 x 64 per wave)
        v8f_u s[4] = {};
#pragma unroll
        for (int t = 0; t < 4; ++t)
#pragma unroll
            for (int c = 0; c < 4; ++c) {
                v16bf_u bk;
#pragma unroll
                for (int i = 0; i < 16; ++i)
                    bk.e[i] = Ks[cur][t * 16 + ln][c * 32 + 16 * g + i];
                s[t].v = wmma_bf16(aq[c].v, bk.v, s[t].v);
            }

        // online softmax
        float alpha[8];
#pragma unroll
        for (int r = 0; r < 8; ++r) {
            int qg = q0 + w * 16 + r + 8 * g;
            float rm = -1e30f;
#pragma unroll
            for (int t = 0; t < 4; ++t) {
                float f = s[t].e[r] * scale;
                int kg = kblk * 64 + t * 16 + ln;
                if (kg > qg) f = -1e30f;
                s[t].e[r] = f;
                rm = fmaxf(rm, f);
            }
#pragma unroll
            for (int off = 1; off < 16; off <<= 1)
                rm = fmaxf(rm, __shfl_xor(rm, off, 32));
            float mnew = fmaxf(mst[r], rm);
            float rs = 0.f;
#pragma unroll
            for (int t = 0; t < 4; ++t) {
                float p = __expf(s[t].e[r] - mnew);
                rs += p;
                Ps[w][r + 8 * g][t * 16 + ln] = (__bf16)p;
            }
#pragma unroll
            for (int off = 1; off < 16; off <<= 1)
                rs += __shfl_xor(rs, off, 32);
            alpha[r] = __expf(mst[r] - mnew);
            mst[r] = mnew;
            lst[r] = lst[r] * alpha[r] + rs;
        }
#pragma unroll
        for (int t = 0; t < 8; ++t)
#pragma unroll
            for (int r = 0; r < 8; ++r) o[t].e[r] *= alpha[r];

        // wave-local LDS RAW fence before re-reading P in A-layout
        asm volatile("s_wait_dscnt 0" ::: "memory");

        // O += P * V  (16 x 128 per wave)
#pragma unroll
        for (int c = 0; c < 2; ++c) {
            v16bf_u ap;
#pragma unroll
            for (int i = 0; i < 16; ++i)
                ap.e[i] = Ps[w][ln][c * 32 + 16 * g + i];
#pragma unroll
            for (int t = 0; t < 8; ++t) {
                v16bf_u bv;
#pragma unroll
                for (int i = 0; i < 16; ++i)
                    bv.e[i] = Vt[cur][t * 16 + ln][c * 32 + 16 * g + i];
                o[t].v = wmma_bf16(ap.v, bv.v, o[t].v);
            }
        }

        if (more) {
#if HAVE_TDM
            if (w == 0) __builtin_amdgcn_s_wait_tensorcnt(0);
#endif
            __syncthreads();
            cur ^= 1;
        }
    }

    // finalize and store attn output (bf16, [S, NH*D])
#pragma unroll
    for (int t = 0; t < 8; ++t)
#pragma unroll
        for (int r = 0; r < 8; ++r) {
            int qg = q0 + w * 16 + r + 8 * g;
            ob[(size_t)qg * (NH * HD) + (size_t)h * HD + t * 16 + ln] =
                (__bf16)(o[t].e[r] / lst[r]);
        }
}

// ---------------- host-side orchestration ----------------
extern "C" void kernel_launch(void* const* d_in, const int* in_sizes, int n_in,
                              void* d_out, int out_size, void* d_ws, size_t ws_size,
                              hipStream_t stream) {
    const float* x    = (const float*)d_in[0];
    const float* cosb = (const float*)d_in[1];
    const float* sinb = (const float*)d_in[2];
    const float* qw   = (const float*)d_in[3];
    const float* kw   = (const float*)d_in[4];
    const float* vw   = (const float*)d_in[5];
    const float* ow   = (const float*)d_in[6];
    const float* qnw  = (const float*)d_in[7];
    const float* knw  = (const float*)d_in[8];
    const float* anw  = (const float*)d_in[9];

    char* ws = (char*)d_ws;
    size_t off = 0;
    auto alloc = [&](size_t bytes) { size_t o = off; off += (bytes + 255) & ~(size_t)255; return o; };

    __bf16* xn  = (__bf16*)(ws + alloc((size_t)S_LEN * HID * 2));
    __bf16* wqb = (__bf16*)(ws + alloc((size_t)NH  * HD * HID * 2));
    __bf16* wkb = (__bf16*)(ws + alloc((size_t)NKV * HD * HID * 2));
    __bf16* wvb = (__bf16*)(ws + alloc((size_t)NKV * HD * HID * 2));
    __bf16* wob = (__bf16*)(ws + alloc((size_t)HID * NH * HD * 2));
    float*  qf  = (float*) (ws + alloc((size_t)S_LEN * NH  * HD * 4));
    float*  kf  = (float*) (ws + alloc((size_t)S_LEN * NKV * HD * 4));
    float*  vf  = (float*) (ws + alloc((size_t)S_LEN * NKV * HD * 4));
    __bf16* qbb = (__bf16*)(ws + alloc((size_t)S_LEN * NH  * HD * 2));
    __bf16* kbb = (__bf16*)(ws + alloc((size_t)S_LEN * NKV * HD * 2));
    __bf16* vbb = (__bf16*)(ws + alloc((size_t)S_LEN * NKV * HD * 2));
    __bf16* ab  = (__bf16*)(ws + alloc((size_t)S_LEN * NH  * HD * 2));

    // 1. weight conversion fp32 -> bf16
    {
        int n;
        n = NH * HD * HID;  f32_to_bf16<<<(n + 255) / 256, 256, 0, stream>>>(qw, wqb, n);
        n = NKV * HD * HID; f32_to_bf16<<<(n + 255) / 256, 256, 0, stream>>>(kw, wkb, n);
        n = NKV * HD * HID; f32_to_bf16<<<(n + 255) / 256, 256, 0, stream>>>(vw, wvb, n);
        n = HID * NH * HD;  f32_to_bf16<<<(n + 255) / 256, 256, 0, stream>>>(ow, wob, n);
    }

    // 2. input RMSNorm -> bf16
    rmsnorm_x<<<S_LEN, 256, 0, stream>>>(x, anw, xn);

    // 3. Q/K/V projections (WMMA GEMM)
    gemm_bf16_f32<<<dim3((NH  * HD) / GBN, S_LEN / GBM), 256, 0, stream>>>(xn, wqb, qf, S_LEN, NH  * HD, HID);
    gemm_bf16_f32<<<dim3((NKV * HD) / GBN, S_LEN / GBM), 256, 0, stream>>>(xn, wkb, kf, S_LEN, NKV * HD, HID);
    gemm_bf16_f32<<<dim3((NKV * HD) / GBN, S_LEN / GBM), 256, 0, stream>>>(xn, wvb, vf, S_LEN, NKV * HD, HID);

    // 4. per-head RMSNorm + RoPE (Q, K); V -> bf16
    qknorm_rope<<<dim3(S_LEN, NH),  128, 0, stream>>>(qf, qbb, qnw, cosb, sinb, NH);
    qknorm_rope<<<dim3(S_LEN, NKV), 128, 0, stream>>>(kf, kbb, knw, cosb, sinb, NKV);
    {
        int n = S_LEN * NKV * HD;
        f32_to_bf16<<<(n + 255) / 256, 256, 0, stream>>>(vf, vbb, n);
    }

    // 5. causal GQA flash attention (WMMA + TDM)
    flash_attn<<<dim3(S_LEN / 64, NH), 128, 0, stream>>>(qbb, kbb, vbb, ab);

    // 6. output projection -> fp32 d_out
    gemm_bf16_f32<<<dim3(HID / GBN, S_LEN / GBM), 256, 0, stream>>>(ab, wob, (float*)d_out, S_LEN, HID, NH * HD);
}
